// UltimateNet_21148418965764
// MI455X (gfx1250) — compile-verified, matrix operand order checked by reading, not run on hardware
//
#include <hip/hip_runtime.h>
#include <hip/hip_bf16.h>

typedef __attribute__((ext_vector_type(16))) _Float16 v16h;
typedef __attribute__((ext_vector_type(8)))  float    v8f;

#define F_CNT 64
#define U_CNT 2048
#define SEQ   4224          // 64 + 2048 + 64 + 2048
#define S0    64
#define S1    2112
#define S2    2176

// ---------------------------------------------------------------------------
// Spatial convs (board path)
// ---------------------------------------------------------------------------
__global__ void conv3x3_kernel(const float* __restrict__ in, const float* __restrict__ w,
                               const float* __restrict__ b, const float* __restrict__ alpha,
                               float* __restrict__ out, int Cin, int Cout, int H, int W,
                               int use_prelu) {
  int idx = blockIdx.x * blockDim.x + threadIdx.x;
  int n = Cout * H * W;
  if (idx >= n) return;
  int oc = idx / (H * W);
  int rem = idx - oc * H * W;
  int y = rem / W, x = rem - (rem / W) * W;
  float acc = b[oc];
  for (int ic = 0; ic < Cin; ++ic) {
    const float* ip = in + (size_t)ic * H * W;
    const float* wp = w + ((size_t)oc * Cin + ic) * 9;
    for (int ky = 0; ky < 3; ++ky) {
      int iy = y + ky - 1;
      if (iy < 0 || iy >= H) continue;
      for (int kx = 0; kx < 3; ++kx) {
        int ix = x + kx - 1;
        if (ix < 0 || ix >= W) continue;
        acc += wp[ky * 3 + kx] * ip[iy * W + ix];
      }
    }
  }
  if (use_prelu) { float a = alpha[0]; acc = acc > 0.f ? acc : a * acc; }
  out[idx] = acc;
}

// stride-3, pad-0 conv: 48x48 -> 16x16
__global__ void conv3x3_s3_kernel(const float* __restrict__ in, const float* __restrict__ w,
                                  const float* __restrict__ b, float* __restrict__ out,
                                  int Cin, int Cout) {
  int idx = blockIdx.x * blockDim.x + threadIdx.x;
  int n = Cout * 16 * 16;
  if (idx >= n) return;
  int oc = idx / 256;
  int rem = idx - oc * 256;
  int oy = rem / 16, ox = rem - (rem / 16) * 16;
  float acc = b[oc];
  for (int ic = 0; ic < Cin; ++ic) {
    const float* ip = in + (size_t)ic * 48 * 48;
    const float* wp = w + ((size_t)oc * Cin + ic) * 9;
    for (int ky = 0; ky < 3; ++ky)
      for (int kx = 0; kx < 3; ++kx)
        acc += wp[ky * 3 + kx] * ip[(oy * 3 + ky) * 48 + (ox * 3 + kx)];
  }
  out[idx] = acc;
}

// ---------------------------------------------------------------------------
// Gather spatial features at positions, concat to token features
// xfeat: [4224][40] (fac: 7+24=31 valid, unit: 14+24=38 valid, rest zero)
// ---------------------------------------------------------------------------
__global__ void gather_concat_kernel(const float* __restrict__ f0f, const float* __restrict__ u0f,
                                     const float* __restrict__ f1f, const float* __restrict__ u1f,
                                     const int* __restrict__ f0p, const int* __restrict__ u0p,
                                     const int* __restrict__ f1p, const int* __restrict__ u1p,
                                     const float* __restrict__ sm, float* __restrict__ xfeat) {
  int t = blockIdx.x * blockDim.x + threadIdx.x;
  if (t >= SEQ) return;
  const float* feat; const int* pos; int fdim, row;
  if (t < S0)       { feat = f0f; pos = f0p; fdim = 7;  row = t; }
  else if (t < S1)  { feat = u0f; pos = u0p; fdim = 14; row = t - S0; }
  else if (t < S2)  { feat = f1f; pos = f1p; fdim = 7;  row = t - S1; }
  else              { feat = u1f; pos = u1p; fdim = 14; row = t - S2; }
  float* xr = xfeat + (size_t)t * 40;
  for (int c = 0; c < fdim; ++c) xr[c] = feat[(size_t)row * fdim + c];
  int px = pos[row * 2 + 0], py = pos[row * 2 + 1];
  for (int c = 0; c < 24; ++c) xr[fdim + c] = sm[c * 256 + px * 16 + py];
  for (int c = fdim + 24; c < 40; ++c) xr[c] = 0.f;
}

// ---------------------------------------------------------------------------
// WMMA QKV projections: one wave per 16-token tile.
// A = token tile (16 x 32*NCH f16, zero-padded past indim), B = weight tile.
// Emits 2 heads x 3 (QKV) x 2 N-tiles x NCH WMMAs per tile.
// Segment boundaries (64/2112/2176) are all multiples of 16 -> uniform waves.
// ---------------------------------------------------------------------------
template <int NCH>
__global__ __launch_bounds__(32)
void qkv_proj_wmma_kernel(const float* __restrict__ xin, int xstride,
                          int fac_dim, int unit_dim,
                          const float* __restrict__ fw, const float* __restrict__ fb,
                          const float* __restrict__ uw, const float* __restrict__ ub,
                          _Float16* __restrict__ Qo, _Float16* __restrict__ Ko,
                          _Float16* __restrict__ Vo) {
  const int tbase = blockIdx.x * 16;
  const int lane  = threadIdx.x;
  const int hi = lane >> 4, lid = lane & 15;
  const int kc0 = hi ? 8 : 0, kc1 = hi ? 24 : 16;

  int player, isfac;
  if (tbase < S0)       { player = 0; isfac = 1; }
  else if (tbase < S1)  { player = 0; isfac = 0; }
  else if (tbase < S2)  { player = 1; isfac = 1; }
  else                  { player = 1; isfac = 0; }
  const int indim = isfac ? fac_dim : unit_dim;
  const float* Wm = isfac ? fw : uw;
  const float* Bs = isfac ? fb : ub;

  // A operands (row = tbase + lid), one per 32-wide K chunk
  v16h a[NCH];
  {
    const float* xr = xin + (size_t)(tbase + lid) * xstride;
    for (int c = 0; c < NCH; ++c)
      for (int j = 0; j < 8; ++j) {
        int k0 = c * 32 + kc0 + j, k1 = c * 32 + kc1 + j;
        a[c][j]     = (k0 < indim) ? (_Float16)xr[k0] : (_Float16)0.f;
        a[c][8 + j] = (k1 < indim) ? (_Float16)xr[k1] : (_Float16)0.f;
      }
  }

  for (int h = 0; h < 2; ++h) {
    int b0 = h * 6 + player * 3;
    for (int qkv = 0; qkv < 3; ++qkv) {
      const float* Wp = Wm + (size_t)(b0 + qkv) * 32 * indim;
      const float* bp = Bs + (size_t)(b0 + qkv) * 32;
      _Float16* dst = (qkv == 0) ? Qo : ((qkv == 1) ? Ko : Vo);
      for (int nt = 0; nt < 2; ++nt) {
        int n = nt * 16 + lid;
        v8f acc = {};
        for (int c = 0; c < NCH; ++c) {
          v16h bop;
          for (int j = 0; j < 8; ++j) {
            int k0 = c * 32 + kc0 + j, k1 = c * 32 + kc1 + j;
            bop[j]     = (k0 < indim) ? (_Float16)Wp[(size_t)n * indim + k0] : (_Float16)0.f;
            bop[8 + j] = (k1 < indim) ? (_Float16)Wp[(size_t)n * indim + k1] : (_Float16)0.f;
          }
          acc = __builtin_amdgcn_wmma_f32_16x16x32_f16(false, a[c], false, bop,
                                                       (short)0, acc, false, false);
        }
        float bias = bp[n];
        for (int i = 0; i < 8; ++i) {
          int t = tbase + i + 8 * hi;
          dst[((size_t)h * SEQ + t) * 32 + n] = (_Float16)(acc[i] + bias);
        }
      }
    }
  }
}

// ---------------------------------------------------------------------------
// Flash attention with WMMA:  one wave per (16-row Q tile, head)
// K=32 exactly matches v_wmma_f32_16x16x32_f16.  Per 32-kv block:
//   2 WMMA (scores) + online softmax + 2 WMMA (P·V)
// ---------------------------------------------------------------------------
__global__ __launch_bounds__(32)
void attn_flash_kernel(const _Float16* __restrict__ Qb, const _Float16* __restrict__ Kb,
                       const _Float16* __restrict__ Vb, float* __restrict__ Ob, int seqlen) {
  __shared__ _Float16 Pl[16][32];   // P tile staging (D-layout -> A-layout transpose)
  __shared__ _Float16 Vt[32][32];   // V tile staging for column gather

  const int qtile = blockIdx.x;
  const int head  = blockIdx.y;
  const int lane  = threadIdx.x;
  const _Float16* Q  = Qb + (size_t)head * seqlen * 32;
  const _Float16* Kp = Kb + (size_t)head * seqlen * 32;
  const _Float16* Vp = Vb + (size_t)head * seqlen * 32;
  float* O = Ob + (size_t)head * seqlen * 32;

  const int hi  = lane >> 4;       // half-wave group
  const int lid = lane & 15;
  const int kc0 = hi ? 8 : 0;      // A/B operand K-chunk bases per ISA layout
  const int kc1 = hi ? 24 : 16;

  // Q tile as A operand (16x32 f16)
  v16h a_q;
  {
    const _Float16* qrow = Q + (size_t)(qtile * 16 + lid) * 32;
    for (int j = 0; j < 8; ++j) { a_q[j] = qrow[kc0 + j]; a_q[8 + j] = qrow[kc1 + j]; }
  }

  v8f o_lo = {}; v8f o_hi = {};
  float m_run[8], l_run[8];
  for (int i = 0; i < 8; ++i) { m_run[i] = -1e30f; l_run[i] = 0.f; }

  const float scale = 0.17677669529663687f;   // 1/sqrt(32)
  const int nblk = seqlen >> 5;

  for (int blk = 0; blk < nblk; ++blk) {
    const int kv0 = blk << 5;
    if (blk + 1 < nblk) {
      __builtin_prefetch(Kp + (size_t)(kv0 + 32) * 32, 0, 1);   // global_prefetch_b8
      __builtin_prefetch(Vp + (size_t)(kv0 + 32) * 32, 0, 1);
    }

    // stage V tile (32 kv x 32 d) into LDS, one row per lane
    {
      const _Float16* vr = Vp + (size_t)(kv0 + lane) * 32;
      for (int j = 0; j < 32; ++j) Vt[lane][j] = vr[j];
    }
    __syncthreads();

    // B operands for K^T (32x16 each)
    v16h b_lo, b_hi;
    {
      const _Float16* kr0 = Kp + (size_t)(kv0 + lid) * 32;
      const _Float16* kr1 = Kp + (size_t)(kv0 + 16 + lid) * 32;
      for (int j = 0; j < 8; ++j) { b_lo[j] = kr0[kc0 + j]; b_lo[8 + j] = kr0[kc1 + j]; }
      for (int j = 0; j < 8; ++j) { b_hi[j] = kr1[kc0 + j]; b_hi[8 + j] = kr1[kc1 + j]; }
    }

    v8f s_lo = {}; v8f s_hi = {};
    s_lo = __builtin_amdgcn_wmma_f32_16x16x32_f16(false, a_q, false, b_lo, (short)0, s_lo, false, false);
    s_hi = __builtin_amdgcn_wmma_f32_16x16x32_f16(false, a_q, false, b_hi, (short)0, s_hi, false, false);

    // online softmax: rows live within 16-lane halves, reduce with xor shuffles
    float p_lo[8], p_hi[8];
    for (int i = 0; i < 8; ++i) {
      float slo = s_lo[i] * scale, shi = s_hi[i] * scale;
      float mx = fmaxf(slo, shi);
      for (int m = 8; m >= 1; m >>= 1) mx = fmaxf(mx, __shfl_xor(mx, m, 32));
      float m_new = fmaxf(m_run[i], mx);
      float corr = __expf(m_run[i] - m_new);
      o_lo[i] *= corr; o_hi[i] *= corr; l_run[i] *= corr;
      float elo = __expf(slo - m_new), ehi = __expf(shi - m_new);
      float rs = elo + ehi;
      for (int m = 8; m >= 1; m >>= 1) rs += __shfl_xor(rs, m, 32);
      l_run[i] += rs;
      m_run[i] = m_new;
      p_lo[i] = elo; p_hi[i] = ehi;
    }

    // D-layout -> A-layout for P via LDS
    for (int i = 0; i < 8; ++i) {
      int r = i + 8 * hi;
      Pl[r][lid]      = (_Float16)p_lo[i];
      Pl[r][16 + lid] = (_Float16)p_hi[i];
    }
    __syncthreads();

    v16h a_p;
    for (int j = 0; j < 8; ++j) { a_p[j] = Pl[lid][kc0 + j]; a_p[8 + j] = Pl[lid][kc1 + j]; }

    // B operands from V columns (kv-major)
    v16h b_v0, b_v1;
    for (int j = 0; j < 8; ++j) {
      b_v0[j]     = Vt[kc0 + j][lid];
      b_v0[8 + j] = Vt[kc1 + j][lid];
      b_v1[j]     = Vt[kc0 + j][16 + lid];
      b_v1[8 + j] = Vt[kc1 + j][16 + lid];
    }

    o_lo = __builtin_amdgcn_wmma_f32_16x16x32_f16(false, a_p, false, b_v0, (short)0, o_lo, false, false);
    o_hi = __builtin_amdgcn_wmma_f32_16x16x32_f16(false, a_p, false, b_v1, (short)0, o_hi, false, false);
    __syncthreads();
  }

  for (int i = 0; i < 8; ++i) {
    int r = i + 8 * hi;
    float inv = 1.0f / l_run[i];
    O[(size_t)(qtile * 16 + r) * 32 + lid]      = o_lo[i] * inv;
    O[(size_t)(qtile * 16 + r) * 32 + 16 + lid] = o_hi[i] * inv;
  }
}

// ---------------------------------------------------------------------------
// WMMA head-concat output projection, optional residual + layernorm.
// A = concat(head0, head1) (16 x 64, 2 K-chunks); B = lin_w tiles; 4 WMMAs.
// ---------------------------------------------------------------------------
__global__ __launch_bounds__(32)
void out_proj_wmma_kernel(const float* __restrict__ H,   // [2][SEQ][32]
                          const float* __restrict__ lw, const float* __restrict__ lb,
                          const float* __restrict__ ln_g, const float* __restrict__ ln_b,
                          float* __restrict__ seq, int do_res_ln) {
  const int tbase = blockIdx.x * 16;
  const int lane  = threadIdx.x;
  const int hi = lane >> 4, lid = lane & 15;
  const int kc0 = hi ? 8 : 0, kc1 = hi ? 24 : 16;

  v16h a0, a1;
  {
    const float* r0 = H + (size_t)(tbase + lid) * 32;
    const float* r1 = H + (size_t)(SEQ + tbase + lid) * 32;
    for (int j = 0; j < 8; ++j) {
      a0[j] = (_Float16)r0[kc0 + j]; a0[8 + j] = (_Float16)r0[kc1 + j];
      a1[j] = (_Float16)r1[kc0 + j]; a1[8 + j] = (_Float16)r1[kc1 + j];
    }
  }

  v8f acc[2];
  for (int nt = 0; nt < 2; ++nt) {
    int n = nt * 16 + lid;
    const float* wr = lw + (size_t)n * 64;
    v16h b0, b1;
    for (int j = 0; j < 8; ++j) {
      b0[j] = (_Float16)wr[kc0 + j];      b0[8 + j] = (_Float16)wr[kc1 + j];
      b1[j] = (_Float16)wr[32 + kc0 + j]; b1[8 + j] = (_Float16)wr[32 + kc1 + j];
    }
    v8f c = {};
    c = __builtin_amdgcn_wmma_f32_16x16x32_f16(false, a0, false, b0, (short)0, c, false, false);
    c = __builtin_amdgcn_wmma_f32_16x16x32_f16(false, a1, false, b1, (short)0, c, false, false);
    acc[nt] = c;
  }

  float bias0 = lb[lid], bias1 = lb[16 + lid];
  float g0 = 0.f, g1 = 0.f, be0 = 0.f, be1 = 0.f;
  if (do_res_ln) { g0 = ln_g[lid]; g1 = ln_g[16 + lid]; be0 = ln_b[lid]; be1 = ln_b[16 + lid]; }

  for (int i = 0; i < 8; ++i) {
    int t = tbase + i + 8 * hi;
    float v0 = acc[0][i] + bias0;
    float v1 = acc[1][i] + bias1;
    if (do_res_ln) {
      v0 += seq[(size_t)t * 32 + lid];
      v1 += seq[(size_t)t * 32 + 16 + lid];
      float mu = v0 + v1;
      for (int m = 8; m >= 1; m >>= 1) mu += __shfl_xor(mu, m, 32);
      mu *= (1.f / 32.f);
      float c0 = v0 - mu, c1 = v1 - mu;
      float var = c0 * c0 + c1 * c1;
      for (int m = 8; m >= 1; m >>= 1) var += __shfl_xor(var, m, 32);
      var *= (1.f / 32.f);
      float inv = rsqrtf(var + 1e-5f);
      v0 = c0 * inv * g0 + be0;
      v1 = c1 * inv * g1 + be1;
    }
    seq[(size_t)t * 32 + lid]      = v0;
    seq[(size_t)t * 32 + 16 + lid] = v1;
  }
}

// ---------------------------------------------------------------------------
// per-token output heads
// ---------------------------------------------------------------------------
__global__ void heads_kernel(const float* __restrict__ seq,
                             const float* fo_w1, const float* fo_b1, const float* fo_a,
                             const float* fo_w2, const float* fo_b2,
                             const float* ro_w1, const float* ro_b1, const float* ro_a,
                             const float* ro_w2, const float* ro_b2,
                             float* __restrict__ out) {
  __shared__ float hidv[32];
  int t = blockIdx.x, d = threadIdx.x;
  int isfac, base, row;
  if (t < S0)       { isfac = 1; base = 0;     row = t; }
  else if (t < S1)  { isfac = 0; base = 256;   row = t - S0; }
  else if (t < S2)  { isfac = 1; base = 35072; row = t - S1; }
  else              { isfac = 0; base = 35328; row = t - S2; }
  const float *w1, *b1, *a, *w2, *b2; int odim;
  if (isfac) { w1 = fo_w1; b1 = fo_b1; a = fo_a; w2 = fo_w2; b2 = fo_b2; odim = 4; }
  else       { w1 = ro_w1; b1 = ro_b1; a = ro_a; w2 = ro_w2; b2 = ro_b2; odim = 17; }
  const float* x = seq + (size_t)t * 32;
  float acc = b1[d];
  for (int k = 0; k < 32; ++k) acc += w1[d * 32 + k] * x[k];
  float al = a[0];
  hidv[d] = acc > 0.f ? acc : al * acc;
  __syncthreads();
  if (d < odim) {
    float o = b2[d];
    for (int k = 0; k < 32; ++k) o += w2[d * 32 + k] * hidv[k];
    out[base + (size_t)row * odim + d] = o;
  }
}

// ---------------------------------------------------------------------------
// Value path: scatter masks, convs+pool, final linear+tanh
// ---------------------------------------------------------------------------
__global__ void zero_kernel(float* p, int n) {
  int i = blockIdx.x * blockDim.x + threadIdx.x;
  if (i < n) p[i] = 0.f;
}

__global__ void scatter_kernel(const float* __restrict__ feat, const int* __restrict__ pos,
                               int N, int C, int chbase, float* __restrict__ vmap) {
  int i = blockIdx.x * blockDim.x + threadIdx.x;
  if (i >= N * C) return;
  int n = i / C, c = i - n * C;
  int px = pos[n * 2 + 0], py = pos[n * 2 + 1];
  atomicAdd(&vmap[(size_t)(chbase + c) * 256 + px * 16 + py], feat[(size_t)n * C + c]);
}

__global__ void copy_kernel(const float* __restrict__ src, float* __restrict__ dst, int n) {
  int i = blockIdx.x * blockDim.x + threadIdx.x;
  if (i < n) dst[i] = src[i];
}

// 3x3 pad1 conv + prelu + 2x2 maxpool, one thread per pooled output
__global__ void conv_pool_kernel(const float* __restrict__ in, const float* __restrict__ w,
                                 const float* __restrict__ b, const float* __restrict__ alpha,
                                 float* __restrict__ out, int Cin, int Cout, int Hin) {
  int Ho = Hin / 2;
  int idx = blockIdx.x * blockDim.x + threadIdx.x;
  if (idx >= Cout * Ho * Ho) return;
  int oc = idx / (Ho * Ho);
  int rem = idx - oc * Ho * Ho;
  int py = rem / Ho, px = rem - (rem / Ho) * Ho;
  float a = alpha[0];
  float best = -1e30f;
  for (int dy = 0; dy < 2; ++dy)
    for (int dx = 0; dx < 2; ++dx) {
      int y = 2 * py + dy, x = 2 * px + dx;
      float acc = b[oc];
      for (int ic = 0; ic < Cin; ++ic) {
        const float* ip = in + (size_t)ic * Hin * Hin;
        const float* wp = w + ((size_t)oc * Cin + ic) * 9;
        for (int ky = 0; ky < 3; ++ky) {
          int iy = y + ky - 1;
          if (iy < 0 || iy >= Hin) continue;
          for (int kx = 0; kx < 3; ++kx) {
            int ix = x + kx - 1;
            if (ix < 0 || ix >= Hin) continue;
            acc += wp[ky * 3 + kx] * ip[iy * Hin + ix];
          }
        }
      }
      acc = acc > 0.f ? acc : a * acc;
      best = fmaxf(best, acc);
    }
  out[idx] = best;
}

__global__ void value_head_kernel(const float* __restrict__ vin, const float* __restrict__ lw,
                                  const float* __restrict__ lb, float* __restrict__ out) {
  __shared__ float red[256];
  int d = threadIdx.x;
  float acc = 0.f;
  for (int k = d; k < 2048; k += 256) acc += vin[k] * lw[k];
  red[d] = acc;
  __syncthreads();
  for (int s = 128; s > 0; s >>= 1) { if (d < s) red[d] += red[d + s]; __syncthreads(); }
  if (d == 0) out[0] = tanhf(red[0] + lb[0]);
}

// ---------------------------------------------------------------------------
extern "C" void kernel_launch(void* const* d_in, const int* in_sizes, int n_in,
                              void* d_out, int out_size, void* d_ws, size_t ws_size,
                              hipStream_t stream) {
  (void)in_sizes; (void)n_in; (void)out_size; (void)ws_size;

  const float* board   = (const float*)d_in[0];
  const float* f0f     = (const float*)d_in[1];
  const float* u0f     = (const float*)d_in[2];
  const float* f1f     = (const float*)d_in[3];
  const float* u1f     = (const float*)d_in[4];
  const float* sp_w1   = (const float*)d_in[5];
  const float* sp_b1   = (const float*)d_in[6];
  const float* sp_a1   = (const float*)d_in[7];
  const float* sp_w2   = (const float*)d_in[8];
  const float* sp_b2   = (const float*)d_in[9];
  const float* sp_a2   = (const float*)d_in[10];
  const float* sp_w3   = (const float*)d_in[11];
  const float* sp_b3   = (const float*)d_in[12];
  const float* enc_fw  = (const float*)d_in[13];
  const float* enc_fb  = (const float*)d_in[14];
  const float* enc_uw  = (const float*)d_in[15];
  const float* enc_ub  = (const float*)d_in[16];
  const float* enc_lw  = (const float*)d_in[17];
  const float* enc_lb  = (const float*)d_in[18];
  const float* ext_fw  = (const float*)d_in[19];
  const float* ext_fb  = (const float*)d_in[20];
  const float* ext_uw  = (const float*)d_in[21];
  const float* ext_ub  = (const float*)d_in[22];
  const float* ext_lw  = (const float*)d_in[23];
  const float* ext_lb  = (const float*)d_in[24];
  const float* ext_lng = (const float*)d_in[25];
  const float* ext_lnb = (const float*)d_in[26];
  const float* v_w1    = (const float*)d_in[27];
  const float* v_b1    = (const float*)d_in[28];
  const float* v_a1    = (const float*)d_in[29];
  const float* v_w2    = (const float*)d_in[30];
  const float* v_b2    = (const float*)d_in[31];
  const float* v_a2    = (const float*)d_in[32];
  const float* v_lw    = (const float*)d_in[33];
  const float* v_lb    = (const float*)d_in[34];
  const float* fo_w1   = (const float*)d_in[35];
  const float* fo_b1   = (const float*)d_in[36];
  const float* fo_a    = (const float*)d_in[37];
  const float* fo_w2   = (const float*)d_in[38];
  const float* fo_b2   = (const float*)d_in[39];
  const float* ro_w1   = (const float*)d_in[40];
  const float* ro_b1   = (const float*)d_in[41];
  const float* ro_a    = (const float*)d_in[42];
  const float* ro_w2   = (const float*)d_in[43];
  const float* ro_b2   = (const float*)d_in[44];
  const int*   f0p     = (const int*)d_in[45];
  const int*   u0p     = (const int*)d_in[46];
  const int*   f1p     = (const int*)d_in[47];
  const int*   u1p     = (const int*)d_in[48];

  float* out = (float*)d_out;

  // workspace bump allocator
  char* W = (char*)d_ws;
  size_t off = 0;
  auto alloc = [&](size_t bytes) -> void* {
    off = (off + 255) & ~(size_t)255;
    void* p = W + off;
    off += bytes;
    return p;
  };
  float*    conv1o = (float*)alloc((size_t)48 * 48 * 48 * 4);
  float*    conv2o = (float*)alloc((size_t)96 * 48 * 48 * 4);
  float*    sm     = (float*)alloc((size_t)24 * 256 * 4);
  float*    xfeat  = (float*)alloc((size_t)SEQ * 40 * 4);
  float*    seq    = (float*)alloc((size_t)SEQ * 32 * 4);
  _Float16* Qh     = (_Float16*)alloc((size_t)2 * SEQ * 32 * 2);
  _Float16* Kh     = (_Float16*)alloc((size_t)2 * SEQ * 32 * 2);
  _Float16* Vh     = (_Float16*)alloc((size_t)2 * SEQ * 32 * 2);
  float*    attnO  = (float*)alloc((size_t)2 * SEQ * 32 * 4);
  float*    vmap   = (float*)alloc((size_t)66 * 256 * 4);
  float*    vc1    = (float*)alloc((size_t)96 * 64 * 4);
  float*    vc2    = (float*)alloc((size_t)128 * 16 * 4);

  // --- spatial conv stack ---
  {
    int n1 = 48 * 48 * 48;
    conv3x3_kernel<<<(n1 + 255) / 256, 256, 0, stream>>>(board, sp_w1, sp_b1, sp_a1,
                                                         conv1o, 14, 48, 48, 48, 1);
    int n2 = 96 * 48 * 48;
    conv3x3_kernel<<<(n2 + 255) / 256, 256, 0, stream>>>(conv1o, sp_w2, sp_b2, sp_a2,
                                                         conv2o, 48, 96, 48, 48, 1);
    int n3 = 24 * 256;
    conv3x3_s3_kernel<<<(n3 + 255) / 256, 256, 0, stream>>>(conv2o, sp_w3, sp_b3, sm, 96, 24);
  }

  // --- token features ---
  gather_concat_kernel<<<(SEQ + 255) / 256, 256, 0, stream>>>(f0f, u0f, f1f, u1f,
                                                              f0p, u0p, f1p, u1p, sm, xfeat);

  // --- encoder layer (layer 0): indim 31/38 -> 2 K-chunks ---
  qkv_proj_wmma_kernel<2><<<SEQ / 16, 32, 0, stream>>>(xfeat, 40, 31, 38,
                                                       enc_fw, enc_fb, enc_uw, enc_ub,
                                                       Qh, Kh, Vh);
  attn_flash_kernel<<<dim3(SEQ / 16, 2), 32, 0, stream>>>(Qh, Kh, Vh, attnO, SEQ);
  out_proj_wmma_kernel<<<SEQ / 16, 32, 0, stream>>>(attnO, enc_lw, enc_lb,
                                                    nullptr, nullptr, seq, 0);

  // --- 2 extra layers with residual + LN: indim 32 -> 1 K-chunk ---
  for (int l = 0; l < 2; ++l) {
    qkv_proj_wmma_kernel<1><<<SEQ / 16, 32, 0, stream>>>(seq, 32, 32, 32,
                                                         ext_fw + (size_t)l * 12 * 32 * 32,
                                                         ext_fb + (size_t)l * 12 * 32,
                                                         ext_uw + (size_t)l * 12 * 32 * 32,
                                                         ext_ub + (size_t)l * 12 * 32,
                                                         Qh, Kh, Vh);
    attn_flash_kernel<<<dim3(SEQ / 16, 2), 32, 0, stream>>>(Qh, Kh, Vh, attnO, SEQ);
    out_proj_wmma_kernel<<<SEQ / 16, 32, 0, stream>>>(attnO,
                                                      ext_lw + (size_t)l * 32 * 64,
                                                      ext_lb + (size_t)l * 32,
                                                      ext_lng + (size_t)l * 32,
                                                      ext_lnb + (size_t)l * 32, seq, 1);
  }

  // --- per-token heads ---
  heads_kernel<<<SEQ, 32, 0, stream>>>(seq, fo_w1, fo_b1, fo_a, fo_w2, fo_b2,
                                       ro_w1, ro_b1, ro_a, ro_w2, ro_b2, out);

  // --- value path ---
  {
    int nz = 66 * 256;
    zero_kernel<<<(nz + 255) / 256, 256, 0, stream>>>(vmap, nz);
    scatter_kernel<<<(U_CNT * 14 + 255) / 256, 256, 0, stream>>>(u1f, u1p, U_CNT, 14, 0, vmap);
    scatter_kernel<<<(F_CNT * 7 + 255) / 256, 256, 0, stream>>>(f1f, f1p, F_CNT, 7, 14, vmap);
    scatter_kernel<<<(U_CNT * 14 + 255) / 256, 256, 0, stream>>>(u0f, u0p, U_CNT, 14, 21, vmap);
    scatter_kernel<<<(F_CNT * 7 + 255) / 256, 256, 0, stream>>>(f0f, f0p, F_CNT, 7, 35, vmap);
    copy_kernel<<<(24 * 256 + 255) / 256, 256, 0, stream>>>(sm, vmap + (size_t)42 * 256, 24 * 256);

    int nv1 = 96 * 64;
    conv_pool_kernel<<<(nv1 + 255) / 256, 256, 0, stream>>>(vmap, v_w1, v_b1, v_a1, vc1, 66, 96, 16);
    int nv2 = 128 * 16;
    conv_pool_kernel<<<(nv2 + 255) / 256, 256, 0, stream>>>(vc1, v_w2, v_b2, v_a2, vc2, 96, 128, 8);
    value_head_kernel<<<1, 256, 0, stream>>>(vc2, v_lw, v_lb, out + 70144);
  }
}
